// Seq2Seq_5342939316683
// MI455X (gfx1250) — matrix operand order repaired and checked by reading, not compiled, not run on hardware
//
#include <hip/hip_runtime.h>
#include <hip/hip_bf16.h>

#define Hh 256
#define Ee 256
#define Bb 64
#define SRCLEN 48
#define TRGLEN 32
#define VTGT 32000
#define G3 768  // 3*H

#define MBLK 128   // logits GEMM: M rows per block (staged in LDS)
#define NBLK 128   // logits GEMM: N cols per block (8 waves x 16)

typedef __attribute__((ext_vector_type(16))) __bf16 v16bf;
typedef __attribute__((ext_vector_type(8)))  float  v8f;

union Frag {
    v16bf v;
    uint4 q[2];
    __device__ Frag() {}
};

__device__ __forceinline__ unsigned short f2bf(float f) {
    unsigned int u = __float_as_uint(f);
    unsigned int r = u + 0x7fffu + ((u >> 16) & 1u);   // round-to-nearest-even
    return (unsigned short)(r >> 16);
}

__device__ __forceinline__ float sigm(float x) {
    return 1.0f / (1.0f + __expf(-x));
}

// ---------------- prep: fp32 -> bf16 convert ----------------
__global__ void cvt_bf16(const float* __restrict__ in, unsigned short* __restrict__ out, int n) {
    int i = blockIdx.x * blockDim.x + threadIdx.x;
    if (i < n) out[i] = f2bf(in[i]);
}

// gather encoder embedded rows: out[row*E + k] = bf16(emb[tok[row]*E + k]), row = b*SRCLEN + t
__global__ void gather_enc(const int* __restrict__ tok, const float* __restrict__ emb,
                           unsigned short* __restrict__ out, int nrows) {
    int i = blockIdx.x * blockDim.x + threadIdx.x;
    if (i < nrows * Ee) {
        int row = i >> 8, k = i & 255;
        out[i] = f2bf(emb[tok[row] * Ee + k]);
    }
}

// decoder teacher forcing: token = (t==0) ? 1 : trg[b*TRGLEN + t-1]
__global__ void gather_dec(const int* __restrict__ trg, const float* __restrict__ emb,
                           unsigned short* __restrict__ out) {
    int i = blockIdx.x * blockDim.x + threadIdx.x;
    if (i < Bb * TRGLEN * Ee) {
        int row = i >> 8, k = i & 255;
        int b = row / TRGLEN, t = row % TRGLEN;
        int tok = (t == 0) ? 1 : trg[b * TRGLEN + t - 1];
        out[i] = f2bf(emb[tok * Ee + k]);
    }
}

// ---------------- xg = A(bf16)[rows,256] @ W(bf16)[768,256]^T + bias, fp32 out ----------------
__global__ void __launch_bounds__(256)
xg_gemm(const unsigned short* __restrict__ A, const unsigned short* __restrict__ W,
        const float* __restrict__ bias, float* __restrict__ out, int mtiles) {
    const int wid  = threadIdx.x >> 5;
    const int lane = threadIdx.x & 31;
    const int tile = blockIdx.x * 8 + wid;
    const int ntiles = G3 / 16;                 // 48
    if (tile >= mtiles * ntiles) return;        // wave-uniform
    const int mt = tile / ntiles, nt = tile % ntiles;
    const int lh = lane & 15;
    const int hi = lane >> 4;                   // 0 or 1
    const int m  = mt * 16 + lh;
    const int n  = nt * 16 + lh;

    v8f acc = {};
    #pragma unroll
    for (int kc = 0; kc < 256; kc += 32) {
        Frag a, b;
        const uint4* ap = (const uint4*)(A + m * 256 + kc + hi * 8);
        a.q[0] = ap[0];                         // K +0..7  (lane<16) / +8..15
        a.q[1] = ap[2];                         // K +16..23         / +24..31
        const uint4* bp = (const uint4*)(W + n * 256 + kc + hi * 8);
        b.q[0] = bp[0];
        b.q[1] = bp[2];                         // same K order as A
        acc = __builtin_amdgcn_wmma_f32_16x16x32_bf16(false, a.v, false, b.v,
                                                      (short)0, acc, false, false);
    }
    const float bn = bias[n];
    #pragma unroll
    for (int v = 0; v < 8; v++) {
        int mm = mt * 16 + v + hi * 8;
        out[(size_t)mm * G3 + n] = acc[v] + bn;
    }
}

// ---------------- sequential GRU recurrence: 1 block, 512 threads = 16 waves ----------------
// h kept fp32 in owning lane's registers + bf16 in double-buffered LDS (A operand of h@Whh^T)
__global__ void __launch_bounds__(512, 1)
gru_rec(const float* __restrict__ xg,            // [B, T, 768] rows = b*T+t
        const unsigned short* __restrict__ Whh,  // [768,256] bf16
        const float* __restrict__ bhh,           // [768]
        float* __restrict__ h_final,             // [64,256] f32: out (enc) / in (dec)
        unsigned short* __restrict__ dec_h,      // [B*T,256] bf16 (decoder only)
        int T, int is_dec) {
    __shared__ __align__(16) unsigned short hbf[2][Bb * Hh];   // 64 KB double buffer

    const int tid  = threadIdx.x;
    const int wid  = tid >> 5;
    const int lane = tid & 31;
    const int lh   = lane & 15;
    const int hi   = lane >> 4;

    float hreg[4][8];

    // init h and hbf[0]
    if (is_dec) {
        #pragma unroll
        for (int u = 0; u < 4; u++) {
            int unit = wid * 4 + u;
            int mt = unit >> 4, jt = unit & 15;
            int j = jt * 16 + lh;
            #pragma unroll
            for (int v = 0; v < 8; v++) {
                int m = mt * 16 + v + hi * 8;
                float h0 = h_final[m * Hh + j];
                hreg[u][v] = h0;
                hbf[0][m * Hh + j] = f2bf(h0);
            }
        }
    } else {
        #pragma unroll
        for (int u = 0; u < 4; u++)
            #pragma unroll
            for (int v = 0; v < 8; v++) hreg[u][v] = 0.0f;
        for (int i = tid; i < Bb * Hh; i += 512) hbf[0][i] = 0;
    }
    __syncthreads();

    int p = 0;
    for (int t = 0; t < T; t++) {
        #pragma unroll
        for (int u = 0; u < 4; u++) {
            int unit = wid * 4 + u;
            int mt = unit >> 4, jt = unit & 15;
            int j  = jt * 16 + lh;
            int nr = j, nz = Hh + j, nn = 2 * Hh + j;

            v8f ar = {}, az = {}, an = {};
            #pragma unroll
            for (int kc = 0; kc < 256; kc += 32) {
                Frag a, br, bz, bn;
                const uint4* ap = (const uint4*)&hbf[p][(mt * 16 + lh) * Hh + kc + hi * 8];
                a.q[0] = ap[0];
                a.q[1] = ap[2];
                const uint4* rp = (const uint4*)(Whh + nr * 256 + kc + hi * 8);
                br.q[0] = rp[0]; br.q[1] = rp[2];
                const uint4* zp = (const uint4*)(Whh + nz * 256 + kc + hi * 8);
                bz.q[0] = zp[0]; bz.q[1] = zp[2];
                const uint4* np = (const uint4*)(Whh + nn * 256 + kc + hi * 8);
                bn.q[0] = np[0]; bn.q[1] = np[2];
                ar = __builtin_amdgcn_wmma_f32_16x16x32_bf16(false, a.v, false, br.v,
                                                             (short)0, ar, false, false);
                az = __builtin_amdgcn_wmma_f32_16x16x32_bf16(false, a.v, false, bz.v,
                                                             (short)0, az, false, false);
                an = __builtin_amdgcn_wmma_f32_16x16x32_bf16(false, a.v, false, bn.v,
                                                             (short)0, an, false, false);
            }
            const float bh_r = bhh[j], bh_z = bhh[Hh + j], bh_n = bhh[2 * Hh + j];
            #pragma unroll
            for (int v = 0; v < 8; v++) {
                int m = mt * 16 + v + hi * 8;
                const float* xgp = xg + ((size_t)m * T + t) * G3;
                float gr = ar[v] + bh_r;
                float gz = az[v] + bh_z;
                float gn = an[v] + bh_n;
                float r  = sigm(xgp[j] + gr);
                float z  = sigm(xgp[Hh + j] + gz);
                float nt_ = tanhf(xgp[2 * Hh + j] + r * gn);
                float h  = (1.0f - z) * nt_ + z * hreg[u][v];
                hreg[u][v] = h;
                unsigned short hb = f2bf(h);
                hbf[p ^ 1][m * Hh + j] = hb;
                if (is_dec) dec_h[((size_t)m * T + t) * Hh + j] = hb;
            }
        }
        __syncthreads();
        p ^= 1;
    }

    if (!is_dec) {
        #pragma unroll
        for (int u = 0; u < 4; u++) {
            int unit = wid * 4 + u;
            int mt = unit >> 4, jt = unit & 15;
            int j = jt * 16 + lh;
            #pragma unroll
            for (int v = 0; v < 8; v++) {
                int m = mt * 16 + v + hi * 8;
                h_final[m * Hh + j] = hreg[u][v];
            }
        }
    }
}

// ---------------- logits = dec_h(bf16)[2048,256] @ fcW(bf16)[32000,256]^T + fc_b ----------------
// Block = 256 threads (8 waves), macro-tile 128M x 128N.
// A block (128x256 bf16 = 64KB, full K) staged once into LDS via async-to-LDS copy;
// each wave owns one 16-col N strip and sweeps 8 M-tiles, reusing each B fragment 8x.
__global__ void __launch_bounds__(256)
logits_gemm(const unsigned short* __restrict__ A, const unsigned short* __restrict__ W,
            const float* __restrict__ bias, float* __restrict__ out) {
    __shared__ __align__(16) unsigned short Abuf[MBLK * 256];   // 64 KB

    const int tid  = threadIdx.x;
    const int wid  = tid >> 5;
    const int lane = tid & 31;
    const int lh   = lane & 15;
    const int hi   = lane >> 4;

    const int nblocks = VTGT / NBLK;        // 250
    const int mb = blockIdx.x / nblocks;    // 0..15
    const int nb = blockIdx.x % nblocks;

    // ---- stage A block into LDS with gfx1250 async-to-LDS DMA ----
    {
        unsigned lds_base = (unsigned)(unsigned long long)
            (__attribute__((address_space(3))) unsigned short*)Abuf;
        const unsigned short* gA = A + (size_t)mb * MBLK * 256;
        #pragma unroll
        for (int i = tid * 8; i < MBLK * 256; i += 256 * 8) {   // 16 x b128 per thread
            unsigned           lds_addr = lds_base + (unsigned)i * 2u;
            unsigned long long gaddr    = (unsigned long long)(gA + i);
            asm volatile("global_load_async_to_lds_b128 %0, %1, off"
                         :: "v"(lds_addr), "v"(gaddr) : "memory");
        }
        asm volatile("s_wait_asynccnt 0x0" ::: "memory");
    }
    __syncthreads();

    const int n = nb * NBLK + wid * 16 + lh;

    v8f zero = {};
    v8f acc[8];
    #pragma unroll
    for (int mt = 0; mt < 8; mt++) acc[mt] = zero;

    #pragma unroll
    for (int kc = 0; kc < 256; kc += 32) {
        Frag b;
        const uint4* bp = (const uint4*)(W + (size_t)n * 256 + kc + hi * 8);
        b.q[0] = bp[0];
        b.q[1] = bp[2];
        #pragma unroll
        for (int mt = 0; mt < 8; mt++) {
            Frag a;
            const uint4* ap = (const uint4*)&Abuf[(mt * 16 + lh) * 256 + kc + hi * 8];
            a.q[0] = ap[0];
            a.q[1] = ap[2];
            acc[mt] = __builtin_amdgcn_wmma_f32_16x16x32_bf16(false, a.v, false, b.v,
                                                              (short)0, acc[mt], false, false);
        }
    }

    const float bn = bias[n];
    #pragma unroll
    for (int mt = 0; mt < 8; mt++) {
        #pragma unroll
        for (int v = 0; v < 8; v++) {
            int mm = mb * MBLK + mt * 16 + v + hi * 8;
            out[(size_t)mm * VTGT + n] = acc[mt][v] + bn;
        }
    }
}

// ---------------- host ----------------
extern "C" void kernel_launch(void* const* d_in, const int* in_sizes, int n_in,
                              void* d_out, int out_size, void* d_ws, size_t ws_size,
                              hipStream_t stream) {
    const int*   src     = (const int*)d_in[0];
    const int*   trg     = (const int*)d_in[1];
    const float* enc_emb = (const float*)d_in[2];
    const float* enc_Wih = (const float*)d_in[3];
    const float* enc_Whh = (const float*)d_in[4];
    const float* enc_bih = (const float*)d_in[5];
    const float* enc_bhh = (const float*)d_in[6];
    const float* dec_emb = (const float*)d_in[7];
    const float* dec_Wih = (const float*)d_in[8];
    const float* dec_Whh = (const float*)d_in[9];
    const float* dec_bih = (const float*)d_in[10];
    const float* dec_bhh = (const float*)d_in[11];
    const float* fc_W    = (const float*)d_in[12];
    const float* fc_b    = (const float*)d_in[13];
    float* out = (float*)d_out;

    char* ws = (char*)d_ws;
    size_t off = 0;
    auto alloc = [&](size_t bytes) -> void* {
        void* p = ws + off;
        off += (bytes + 255) & ~(size_t)255;
        return p;
    };
    float*          enc_xg    = (float*)alloc((size_t)Bb * SRCLEN * G3 * 4);
    float*          dec_xg    = (float*)alloc((size_t)Bb * TRGLEN * G3 * 4);
    float*          h_final   = (float*)alloc((size_t)Bb * Hh * 4);
    unsigned short* encA      = (unsigned short*)alloc((size_t)Bb * SRCLEN * Ee * 2);
    unsigned short* decA      = (unsigned short*)alloc((size_t)Bb * TRGLEN * Ee * 2);
    unsigned short* encWih_bf = (unsigned short*)alloc((size_t)G3 * Ee * 2);
    unsigned short* encWhh_bf = (unsigned short*)alloc((size_t)G3 * Hh * 2);
    unsigned short* decWih_bf = (unsigned short*)alloc((size_t)G3 * Ee * 2);
    unsigned short* decWhh_bf = (unsigned short*)alloc((size_t)G3 * Hh * 2);
    unsigned short* fcW_bf    = (unsigned short*)alloc((size_t)VTGT * Hh * 2);
    unsigned short* dec_h_bf  = (unsigned short*)alloc((size_t)Bb * TRGLEN * Hh * 2);

    const int TPB = 256;
    // weight conversions
    cvt_bf16<<<(G3 * Ee + TPB - 1) / TPB, TPB, 0, stream>>>(enc_Wih, encWih_bf, G3 * Ee);
    cvt_bf16<<<(G3 * Hh + TPB - 1) / TPB, TPB, 0, stream>>>(enc_Whh, encWhh_bf, G3 * Hh);
    cvt_bf16<<<(G3 * Ee + TPB - 1) / TPB, TPB, 0, stream>>>(dec_Wih, decWih_bf, G3 * Ee);
    cvt_bf16<<<(G3 * Hh + TPB - 1) / TPB, TPB, 0, stream>>>(dec_Whh, decWhh_bf, G3 * Hh);
    cvt_bf16<<<(VTGT * Hh + TPB - 1) / TPB, TPB, 0, stream>>>(fc_W, fcW_bf, VTGT * Hh);
    // embedding gathers
    gather_enc<<<(Bb * SRCLEN * Ee + TPB - 1) / TPB, TPB, 0, stream>>>(src, enc_emb, encA, Bb * SRCLEN);
    gather_dec<<<(Bb * TRGLEN * Ee + TPB - 1) / TPB, TPB, 0, stream>>>(trg, dec_emb, decA);
    // input-gate precomputes (WMMA)
    xg_gemm<<<(Bb * SRCLEN / 16) * (G3 / 16) / 8, TPB, 0, stream>>>(encA, encWih_bf, enc_bih, enc_xg, Bb * SRCLEN / 16);
    xg_gemm<<<(Bb * TRGLEN / 16) * (G3 / 16) / 8, TPB, 0, stream>>>(decA, decWih_bf, dec_bih, dec_xg, Bb * TRGLEN / 16);
    // sequential recurrences (single-workgroup, LDS-resident h, WMMA per step)
    gru_rec<<<1, 512, 0, stream>>>(enc_xg, encWhh_bf, enc_bhh, h_final, (unsigned short*)nullptr, SRCLEN, 0);
    gru_rec<<<1, 512, 0, stream>>>(dec_xg, decWhh_bf, dec_bhh, h_final, dec_h_bf, TRGLEN, 1);
    // final projection: LDS-blocked, async-staged A, store-bandwidth-bound
    logits_gemm<<<(Bb * TRGLEN / MBLK) * (VTGT / NBLK), TPB, 0, stream>>>(dec_h_bf, fcW_bf, fc_b, out);
}
